// FlowAwareGCNEncoder_1391569404372
// MI455X (gfx1250) — compile-verified, matrix operand order checked by reading, not run on hardware
//
#include <hip/hip_runtime.h>
#include <hip/hip_bf16.h>

#define DD 512

typedef __attribute__((ext_vector_type(16))) __bf16 v16bf;
typedef __attribute__((ext_vector_type(8)))  __bf16 v8bf;
typedef __attribute__((ext_vector_type(8)))  float  v8f;

__device__ __forceinline__ unsigned short f32_to_bf16u(float f) {
    unsigned int u = __builtin_bit_cast(unsigned int, f);
    unsigned int r = u + 0x7FFFu + ((u >> 16) & 1u);   // round-to-nearest-even
    return (unsigned short)(r >> 16);
}

__device__ __forceinline__ unsigned int pack_bf16x2(float a, float b) {
    unsigned int ua = __builtin_bit_cast(unsigned int, a);
    unsigned int ub = __builtin_bit_cast(unsigned int, b);
    ua = (ua + 0x7FFFu + ((ua >> 16) & 1u)) >> 16;
    ub = (ub + 0x7FFFu + ((ub >> 16) & 1u)) & 0xFFFF0000u;
    return ua | ub;
}

__device__ __forceinline__ void atomAddF(float* p, float v) {
    unsafeAtomicAdd(p, v);   // native global_atomic_add_f32 on gfx1250
}

// ---------------- fp32 -> bf16 bulk conversion (2 elems / thread) ----------------

__global__ void conv_bf16_kern(const float* __restrict__ src,
                               unsigned short* __restrict__ dst, long long n) {
    long long i = ((long long)blockIdx.x * blockDim.x + threadIdx.x) * 2;
    if (i < n) {
        float2 f = *(const float2*)(src + i);
        ((unsigned int*)dst)[i >> 1] = pack_bf16x2(f.x, f.y);
    }
}

// ---------------- CSR build + normalization precompute ----------------

__global__ void init_kern(float* __restrict__ deg, int* __restrict__ cnt, int n) {
    int i = blockIdx.x * blockDim.x + threadIdx.x;
    if (i < n) { deg[i] = 1.0f; cnt[i] = 0; }    // deg starts at self-loop weight
}

__global__ void deg_cnt_kern(const long long* __restrict__ col,
                             const float* __restrict__ ew,
                             float* __restrict__ deg, int* __restrict__ cnt, int ne) {
    int e = blockIdx.x * blockDim.x + threadIdx.x;
    if (e < ne) {
        int c = (int)col[e];
        atomAddF(&deg[c], ew[e]);
        atomicAdd(&cnt[c], 1);
    }
}

__global__ void deg_inv_kern(float* __restrict__ deg, int n) {
    int i = blockIdx.x * blockDim.x + threadIdx.x;
    if (i < n) {
        float d = deg[i];
        deg[i] = (d > 0.0f) ? rsqrtf(d) : 0.0f;
    }
}

// single-block exclusive scan with carry (n ~ 50000; runs once per call)
__global__ __launch_bounds__(256)
void scan_kern(const int* __restrict__ cnt, int* __restrict__ base, int n) {
    __shared__ int tmp[256];
    __shared__ int carry;
    if (threadIdx.x == 0) carry = 0;
    __syncthreads();
    for (int start = 0; start < n; start += 256) {
        int i = start + (int)threadIdx.x;
        int v = (i < n) ? cnt[i] : 0;
        tmp[threadIdx.x] = v;
        __syncthreads();
        for (int off = 1; off < 256; off <<= 1) {
            int t = (threadIdx.x >= (unsigned)off) ? tmp[threadIdx.x - off] : 0;
            __syncthreads();
            tmp[threadIdx.x] += t;
            __syncthreads();
        }
        if (i < n) base[i] = carry + tmp[threadIdx.x] - v;   // exclusive
        __syncthreads();
        if (threadIdx.x == 255) carry += tmp[255];
        __syncthreads();
    }
    if (threadIdx.x == 0) base[n] = carry;    // total
}

__global__ void curs_init_kern(const int* __restrict__ base, int* __restrict__ curs, int n) {
    int i = blockIdx.x * blockDim.x + threadIdx.x;
    if (i < n) curs[i] = base[i];
}

// scatter edges into destination buckets as packed {src, norm} records
__global__ void csr_build_kern(const long long* __restrict__ row,
                               const long long* __restrict__ col,
                               const float* __restrict__ ew,
                               const float* __restrict__ dinv,
                               int* __restrict__ curs, int2* __restrict__ erec, int ne) {
    int e = blockIdx.x * blockDim.x + threadIdx.x;
    if (e < ne) {
        int r = (int)row[e], c = (int)col[e];
        int pos = atomicAdd(&curs[c], 1);
        float w = dinv[r] * ew[e] * dinv[c];
        erec[pos] = make_int2(r, __builtin_bit_cast(int, w));
    }
}

// ---------------- WMMA GEMM: H[n, d] = sum_k Xb[n, k] * Wb[d, k] (bf16 in, f32 out)

#define LDS_U32_STR 20   // 40 bf16 per row (32 data + 8 pad), in uints

__global__ __launch_bounds__(128)
void gemm_wmma_kern(const unsigned short* __restrict__ Xb,
                    const unsigned short* __restrict__ Wb,
                    float* __restrict__ H, int nrows) {
    __shared__ unsigned int lsA[64 * LDS_U32_STR];
    __shared__ unsigned int lsB[64 * LDS_U32_STR];

    const int m0   = blockIdx.x * 64;
    const int n0   = blockIdx.y * 64;
    const int tid  = threadIdx.x;
    const int lane = tid & 31;
    const int wave = tid >> 5;
    const int wm   = (wave >> 1) * 32;
    const int wn   = (wave & 1) * 32;

    const int r16 = lane & 15;           // row (A) / col (B) within 16
    const int kb  = (lane >> 4) << 3;    // K sub-base: 0 or 8

    const int rr   = tid >> 1;           // staging row 0..63
    const int half = tid & 1;            // 16-elem half of the K slab
    int gra = m0 + rr;
    if (gra >= nrows) gra = nrows - 1;   // clamp: rows independent, stores guarded
    const int grb = n0 + rr;

    v8f acc[2][2] = {};

    for (int k0 = 0; k0 < DD; k0 += 32) {
        const uint4* pa = (const uint4*)(Xb + (size_t)gra * DD + k0 + half * 16);
        const uint4* pb = (const uint4*)(Wb + (size_t)grb * DD + k0 + half * 16);
        uint4 a0 = pa[0], a1 = pa[1];
        uint4 b0 = pb[0], b1 = pb[1];
        uint4* da = (uint4*)&lsA[rr * LDS_U32_STR + half * 8];
        uint4* db = (uint4*)&lsB[rr * LDS_U32_STR + half * 8];
        da[0] = a0; da[1] = a1;
        db[0] = b0; db[1] = b1;
        __syncthreads();

        v16bf afr[2], bfr[2];
        #pragma unroll
        for (int t = 0; t < 2; ++t) {
            const __bf16* ra = (const __bf16*)&lsA[(wm + t * 16 + r16) * LDS_U32_STR];
            const __bf16* rb = (const __bf16*)&lsB[(wn + t * 16 + r16) * LDS_U32_STR];
            v8bf alo = *(const v8bf*)(ra + kb);
            v8bf ahi = *(const v8bf*)(ra + kb + 16);
            v8bf blo = *(const v8bf*)(rb + kb);
            v8bf bhi = *(const v8bf*)(rb + kb + 16);
            afr[t] = __builtin_shufflevector(alo, ahi, 0,1,2,3,4,5,6,7,8,9,10,11,12,13,14,15);
            bfr[t] = __builtin_shufflevector(blo, bhi, 0,1,2,3,4,5,6,7,8,9,10,11,12,13,14,15);
        }

        #pragma unroll
        for (int mi = 0; mi < 2; ++mi)
            #pragma unroll
            for (int ni = 0; ni < 2; ++ni)
                acc[mi][ni] = __builtin_amdgcn_wmma_f32_16x16x32_bf16(
                    false, afr[mi], false, bfr[ni],
                    (short)0, acc[mi][ni], false, false);
        __syncthreads();
    }

    // C/D layout: VGPR v holds M=v (lanes 0-15) or M=v+8 (lanes 16-31); N = lane&15
    const int mhalf = (lane >> 4) << 3;
    const int ncol  = lane & 15;
    const int mb    = m0 + wm + mhalf;
    const int nb    = n0 + wn + ncol;
    if (m0 + 64 <= nrows) {              // full tile: block-uniform, unpredicated
        #pragma unroll
        for (int mi = 0; mi < 2; ++mi)
            #pragma unroll
            for (int ni = 0; ni < 2; ++ni)
                #pragma unroll
                for (int v = 0; v < 8; ++v)
                    H[(size_t)(mb + mi * 16 + v) * DD + (nb + ni * 16)] = acc[mi][ni][v];
    } else {
        #pragma unroll
        for (int mi = 0; mi < 2; ++mi)
            #pragma unroll
            for (int ni = 0; ni < 2; ++ni)
                #pragma unroll
                for (int v = 0; v < 8; ++v) {
                    int gm = mb + mi * 16 + v;
                    if (gm < nrows)
                        H[(size_t)gm * DD + (nb + ni * 16)] = acc[mi][ni][v];
                }
    }
}

// ---------------- fused SpMM + bias + self-loop + LayerNorm + ReLU + residual ----
// One wave per destination node; full 512-wide row lives in 16 regs/lane.
// Gathers of H rows are served by the 192MB L2 (H is fully resident).

__global__ __launch_bounds__(256)
void spmm_ln_kern(const float* __restrict__ H,
                  const int* __restrict__ base,
                  const int2* __restrict__ erec,
                  const float* __restrict__ dinv,
                  const float* __restrict__ bias,
                  const float* __restrict__ gamma, const float* __restrict__ beta,
                  const float* __restrict__ XIN, float* __restrict__ XOUT,
                  unsigned short* __restrict__ XBF,
                  int n, int do_relu, int write_bf) {
    int node = blockIdx.x * 8 + (threadIdx.x >> 5);
    if (node >= n) return;
    int lane = threadIdx.x & 31;

    float4 acc[4];
    // bias + self-loop message
    {
        float dn = dinv[node];
        float sw = dn * dn;
        const float4* hp = (const float4*)(H + (size_t)node * DD);
        const float4* bp = (const float4*)bias;
        #pragma unroll
        for (int k = 0; k < 4; ++k) {
            float4 h = hp[lane + k * 32];
            float4 b = bp[lane + k * 32];
            acc[k].x = b.x + sw * h.x;
            acc[k].y = b.y + sw * h.y;
            acc[k].z = b.z + sw * h.z;
            acc[k].w = b.w + sw * h.w;
        }
    }
    // gather incoming edges
    int e0 = base[node], e1 = base[node + 1];
    for (int e = e0; e < e1; ++e) {
        int2 rec = erec[e];                       // broadcast load (uniform addr)
        int src = rec.x;
        float w = __builtin_bit_cast(float, rec.y);
        const float4* sp = (const float4*)(H + (size_t)src * DD);
        #pragma unroll
        for (int k = 0; k < 4; ++k) {
            float4 v = sp[lane + k * 32];
            acc[k].x += w * v.x;
            acc[k].y += w * v.y;
            acc[k].z += w * v.z;
            acc[k].w += w * v.w;
        }
    }

    // LayerNorm over the wave-distributed row
    float s = 0.0f, sq = 0.0f;
    #pragma unroll
    for (int k = 0; k < 4; ++k) {
        s  += acc[k].x + acc[k].y + acc[k].z + acc[k].w;
        sq += acc[k].x * acc[k].x + acc[k].y * acc[k].y
            + acc[k].z * acc[k].z + acc[k].w * acc[k].w;
    }
    #pragma unroll
    for (int off = 16; off >= 1; off >>= 1) {
        s  += __shfl_xor(s,  off, 32);
        sq += __shfl_xor(sq, off, 32);
    }
    float mu  = s * (1.0f / DD);
    float var = sq * (1.0f / DD) - mu * mu;
    float rs  = rsqrtf(var + 1e-5f);

    const float4* gp = (const float4*)gamma;
    const float4* tp = (const float4*)beta;
    const float4* xp = (const float4*)(XIN + (size_t)node * DD);
    float4*       op = (float4*)(XOUT + (size_t)node * DD);
    unsigned int* bf = (unsigned int*)(XBF + (size_t)node * DD);
    #pragma unroll
    for (int k = 0; k < 4; ++k) {
        int i4 = lane + k * 32;
        float4 g = gp[i4], t = tp[i4], xi = xp[i4];
        float4 o;
        o.x = (acc[k].x - mu) * rs * g.x + t.x;
        o.y = (acc[k].y - mu) * rs * g.y + t.y;
        o.z = (acc[k].z - mu) * rs * g.z + t.z;
        o.w = (acc[k].w - mu) * rs * g.w + t.w;
        if (do_relu) {
            o.x = fmaxf(o.x, 0.0f); o.y = fmaxf(o.y, 0.0f);
            o.z = fmaxf(o.z, 0.0f); o.w = fmaxf(o.w, 0.0f);
        }
        o.x += xi.x; o.y += xi.y; o.z += xi.z; o.w += xi.w;
        op[i4] = o;
        if (write_bf) {
            bf[i4 * 2 + 0] = pack_bf16x2(o.x, o.y);
            bf[i4 * 2 + 1] = pack_bf16x2(o.z, o.w);
        }
    }
}

// ---------------- host orchestration ----------------

extern "C" void kernel_launch(void* const* d_in, const int* in_sizes, int n_in,
                              void* d_out, int out_size, void* d_ws, size_t ws_size,
                              hipStream_t stream) {
    const float*     X   = (const float*)d_in[0];
    const long long* EI  = (const long long*)d_in[1];   // int64 [2, E]
    const float*     EW  = (const float*)d_in[2];
    const float*     Ws  = (const float*)d_in[3];       // [3, 512, 512]
    const float*     Bs  = (const float*)d_in[4];       // [3, 512]
    const float*     Gms = (const float*)d_in[5];
    const float*     Bts = (const float*)d_in[6];
    float*           OUT = (float*)d_out;

    const int nN = in_sizes[0] / DD;     // 50000
    const int nE = in_sizes[2];          // 800000
    const long long NND = (long long)nN * DD;

    const long long* ROW = EI;
    const long long* COL = EI + nE;

    // workspace layout
    float* H    = (float*)d_ws;                       // [nN, DD] f32
    float* XA   = H + NND;                            // [nN, DD] f32 hidden (in-place LN safe)
    float* DINV = XA + NND;                           // [nN]
    int*   BASE = (int*)(DINV + ((nN + 63) & ~63));   // [nN+1]
    int*   CURS = BASE + ((nN + 64) & ~63);           // [nN]
    int2*  EREC = (int2*)(CURS + ((nN + 63) & ~63));  // [nE] {src, norm}
    unsigned short* XBF = (unsigned short*)(EREC + nE);          // [nN, DD] bf16
    unsigned short* WBF = XBF + NND;                             // [3, DD, DD] bf16

    // ---- bf16 pre-conversion ----
    conv_bf16_kern<<<(int)((NND / 2 + 255) / 256), 256, 0, stream>>>(X, XBF, NND);
    const long long NW = 3LL * DD * DD;
    conv_bf16_kern<<<(int)((NW / 2 + 255) / 256), 256, 0, stream>>>(Ws, WBF, NW);

    // ---- CSR + normalization precompute ----
    init_kern<<<(nN + 255) / 256, 256, 0, stream>>>(DINV, CURS, nN);   // CURS as cnt
    deg_cnt_kern<<<(nE + 255) / 256, 256, 0, stream>>>(COL, EW, DINV, CURS, nE);
    deg_inv_kern<<<(nN + 255) / 256, 256, 0, stream>>>(DINV, nN);
    scan_kern<<<1, 256, 0, stream>>>(CURS, BASE, nN);
    curs_init_kern<<<(nN + 255) / 256, 256, 0, stream>>>(BASE, CURS, nN);
    csr_build_kern<<<(nE + 255) / 256, 256, 0, stream>>>(ROW, COL, EW, DINV, CURS, EREC, nE);

    dim3 ggrid((nN + 63) / 64, DD / 64);
    const int lnBlocks = (nN + 7) / 8;

    for (int l = 0; l < 3; ++l) {
        const float* xin  = (l == 0) ? X : XA;
        float*       xout = (l == 2) ? OUT : XA;
        const unsigned short* Wl = WBF + (size_t)l * DD * DD;

        gemm_wmma_kern<<<ggrid, 128, 0, stream>>>(XBF, Wl, H, nN);
        spmm_ln_kern<<<lnBlocks, 256, 0, stream>>>(
            H, BASE, EREC, DINV, Bs + (size_t)l * DD,
            Gms + (size_t)l * DD, Bts + (size_t)l * DD,
            xin, xout, XBF, nN, (l < 2) ? 1 : 0, (l < 2) ? 1 : 0);
    }
}
